// DimSpecializedAttention_24550033064022
// MI455X (gfx1250) — matrix operand order repaired and checked by reading, not compiled, not run on hardware
//
#include <hip/hip_runtime.h>

#define BATCH 8
#define SEQ 2048
#define DMODEL 1280
#define NHEAD 10
#define HDIM 128

typedef __attribute__((ext_vector_type(16))) __bf16 bf16x16;
typedef __attribute__((ext_vector_type(8)))  float  f32x8;

// ---------------------------------------------------------------------------
// WMMA 16x16x32 bf16 fragment loaders (wave32, layouts per CDNA5 ISA 7.12.2)
// A (16x32): lane l<16 -> row M=l&15, elements 0..7 = K[half*8 ..],
//            elements 8..15 = K[16+half*8 ..]   (two contiguous 8-elem runs)
// B (32x16): same K order as A, N = lane&15
// C/D (16x16 f32): element r -> M = half*8 + r, N = lane&15
// ---------------------------------------------------------------------------

__device__ __forceinline__ bf16x16 load_frag_a(const __bf16* base, int row_stride,
                                               int m0, int k0) {
  int l = threadIdx.x & 31;
  int half = (l >> 4) & 1;
  const __bf16* p = base + (size_t)(m0 + (l & 15)) * row_stride + k0 + half * 8;
  bf16x16 a;
#pragma unroll
  for (int e = 0; e < 8; ++e) a[e] = p[e];
#pragma unroll
  for (int e = 0; e < 8; ++e) a[8 + e] = p[16 + e];
  return a;
}

// B fragment where memory holds the K-dim contiguously per N-row:
// value(k, n) = src[n][k] for row-major src [N x K]. Per lane: two
// contiguous 16-byte runs (K[half*8..], K[16+half*8..]). Used for:
//   scores  B = K^T   (K matrix rows, k-dim = d)
//   weights B[d][e] = W[e][d]  (original row-major W, k-dim = d)
//   PV      B[key][d] = Vt[d][key]  (transposed V, k-dim = key)
__device__ __forceinline__ bf16x16 load_frag_bt(const __bf16* base, int row_stride,
                                                int k0, int n0) {
  int l = threadIdx.x & 31;
  int half = (l >> 4) & 1;
  const __bf16* p = base + (size_t)(n0 + (l & 15)) * row_stride + k0 + half * 8;
  bf16x16 b;
#pragma unroll
  for (int e = 0; e < 8; ++e) b[e] = p[e];
#pragma unroll
  for (int e = 0; e < 8; ++e) b[8 + e] = p[16 + e];
  return b;
}

#define WMMA_BF16(A, B, C) \
  __builtin_amdgcn_wmma_f32_16x16x32_bf16(false, (A), false, (B), (short)0, (C), false, false)

// ---------------------------------------------------------------------------
// bf16 GEMM: C[M x N] = A[M x K] * W^T, W row-major [N x K].
// One 32x64 macro-tile per wave: 2 A-frags x 4 B-frags -> 8 WMMAs per k-step.
// ---------------------------------------------------------------------------
template <bool OUT_F32>
__global__ void __launch_bounds__(256)
gemm_bf16_kernel(const __bf16* __restrict__ A, const __bf16* __restrict__ W,
                 void* __restrict__ Cout, int M, int N, int K) {
  int wave = threadIdx.x >> 5;
  int groups_n = N >> 6;   // 64-wide N groups
  int g = blockIdx.x * 8 + wave;
  if (g >= (M >> 5) * groups_n) return;
  int m0 = (g / groups_n) << 5;
  int n0 = (g % groups_n) << 6;

  f32x8 zero = {};
  f32x8 acc[2][4];
#pragma unroll
  for (int i = 0; i < 2; ++i)
#pragma unroll
    for (int j = 0; j < 4; ++j) acc[i][j] = zero;

  for (int k = 0; k < K; k += 32) {
    bf16x16 a0 = load_frag_a(A, K, m0, k);
    bf16x16 a1 = load_frag_a(A, K, m0 + 16, k);
#pragma unroll
    for (int j = 0; j < 4; ++j) {
      bf16x16 b = load_frag_bt(W, K, k, n0 + j * 16);
      acc[0][j] = WMMA_BF16(a0, b, acc[0][j]);
      acc[1][j] = WMMA_BF16(a1, b, acc[1][j]);
    }
  }

  int l = threadIdx.x & 31;
  int half = (l >> 4) & 1;
#pragma unroll
  for (int i = 0; i < 2; ++i)
#pragma unroll
    for (int j = 0; j < 4; ++j) {
      int col = n0 + j * 16 + (l & 15);
#pragma unroll
      for (int r = 0; r < 8; ++r) {
        size_t idx = (size_t)(m0 + i * 16 + half * 8 + r) * N + col;
        if (OUT_F32) ((float*)Cout)[idx] = acc[i][j][r];
        else         ((__bf16*)Cout)[idx] = (__bf16)acc[i][j][r];
      }
    }
}

// ---------------------------------------------------------------------------
// Flash attention: grid = (SEQ/128, BATCH*NHEAD), 256 thr (8 waves x 16 q-rows)
// V is consumed pre-transposed: vt[b][h][d][t]
// ---------------------------------------------------------------------------
__global__ void __launch_bounds__(256)
attn_kernel(const __bf16* __restrict__ qb, const __bf16* __restrict__ kb,
            const __bf16* __restrict__ vt, const float* __restrict__ gates,
            __bf16* __restrict__ yb) {
  __shared__ __bf16 plds[8][16][32];  // wave-private P staging (C->A relayout)

  int wave = threadIdx.x >> 5;
  int lane = threadIdx.x & 31;
  int half = lane >> 4;
  int ln = lane & 15;
  int bh = blockIdx.y;
  int b = bh / NHEAD, h = bh % NHEAD;
  int qt = blockIdx.x * 128 + wave * 16;  // first q row (t) for this wave

  const size_t bbase = (size_t)b * SEQ * DMODEL + (size_t)h * HDIM;
  const __bf16* Qp = qb + bbase;
  const __bf16* Kp = kb + bbase;
  const __bf16* Vtp = vt + (size_t)(b * NHEAD + h) * HDIM * SEQ;

  bf16x16 qa[4];
#pragma unroll
  for (int kk = 0; kk < 4; ++kk) qa[kk] = load_frag_a(Qp, DMODEL, qt, kk * 32);

  float mrow[8], lrow[8];
  f32x8 zero = {};
  f32x8 o[8];
#pragma unroll
  for (int r = 0; r < 8; ++r) { mrow[r] = -__builtin_inff(); lrow[r] = 0.f; }
#pragma unroll
  for (int j = 0; j < 8; ++j) o[j] = zero;

  const float scale = 0.08838834764831845f;  // 1/sqrt(128)
  const int qhi = qt + 15;

  for (int kt = 0; kt <= qhi; kt += 32) {
    // ---- scores: hoist all 8 K fragments, then 8 back-to-back WMMAs ----
    bf16x16 kf[8];
#pragma unroll
    for (int kk = 0; kk < 4; ++kk) {
      kf[kk]     = load_frag_bt(Kp, DMODEL, kk * 32, kt);
      kf[4 + kk] = load_frag_bt(Kp, DMODEL, kk * 32, kt + 16);
    }
    f32x8 S0 = {}, S1 = {};
#pragma unroll
    for (int kk = 0; kk < 4; ++kk) S0 = WMMA_BF16(qa[kk], kf[kk], S0);
#pragma unroll
    for (int kk = 0; kk < 4; ++kk) S1 = WMMA_BF16(qa[kk], kf[4 + kk], S1);

    // ---- online softmax (row stats across the 16 N-lanes of each half) ----
    int key0 = kt + ln, key1 = kt + 16 + ln;
#pragma unroll
    for (int r = 0; r < 8; ++r) {
      int qr = qt + half * 8 + r;
      float s0 = (key0 <= qr) ? S0[r] * scale : -__builtin_inff();
      float s1 = (key1 <= qr) ? S1[r] * scale : -__builtin_inff();
      float m = fmaxf(s0, s1);
#pragma unroll
      for (int off = 8; off; off >>= 1) m = fmaxf(m, __shfl_xor(m, off, 32));
      float newm = fmaxf(mrow[r], m);
      float alpha = __expf(mrow[r] - newm);
      float p0 = __expf(s0 - newm);
      float p1 = __expf(s1 - newm);
      float rs = p0 + p1;
#pragma unroll
      for (int off = 8; off; off >>= 1) rs += __shfl_xor(rs, off, 32);
      lrow[r] = lrow[r] * alpha + rs;
      mrow[r] = newm;
#pragma unroll
      for (int j = 0; j < 8; ++j) o[j][r] *= alpha;
      plds[wave][half * 8 + r][ln] = (__bf16)p0;
      plds[wave][half * 8 + r][16 + ln] = (__bf16)p1;
    }
    // wave-private LDS write->read: lockstep wave + DScnt drain (CDNA5 counters)
    asm volatile("s_wait_dscnt 0" ::: "memory");
    bf16x16 pa;
#pragma unroll
    for (int e = 0; e < 8; ++e) pa[e] = plds[wave][ln][half * 8 + e];
#pragma unroll
    for (int e = 0; e < 8; ++e) pa[8 + e] = plds[wave][ln][16 + half * 8 + e];

    // ---- O += P * V  (V fragments contiguous via transposed V) ----
#pragma unroll
    for (int j = 0; j < 8; ++j) {
      bf16x16 bv = load_frag_bt(Vtp, SEQ, kt, j * 16);
      o[j] = WMMA_BF16(pa, bv, o[j]);
    }
  }

  float gate = gates[bh];
#pragma unroll
  for (int r = 0; r < 8; ++r) {
    float inv = gate / lrow[r];
    size_t row = (size_t)(qt + half * 8 + r);
#pragma unroll
    for (int j = 0; j < 8; ++j)
      yb[bbase + row * DMODEL + j * 16 + ln] = (__bf16)(o[j][r] * inv);
  }
}

// ---------------------------------------------------------------------------
// Helper kernels
// ---------------------------------------------------------------------------
__global__ void cvt_bf16_kernel(const float* __restrict__ in,
                                __bf16* __restrict__ out, size_t n) {
  size_t i = (size_t)blockIdx.x * blockDim.x + threadIdx.x;
  if (i < n) out[i] = (__bf16)in[i];
}

// LDS-tiled transpose of V [B*T, D] (head-interleaved) -> vt [B][H][hd][T]
__global__ void __launch_bounds__(256)
transpose_v_kernel(const __bf16* __restrict__ v, __bf16* __restrict__ vt) {
  __shared__ __bf16 tile[64][65];
  int bh = blockIdx.z;
  int b = bh / NHEAD, h = bh % NHEAD;
  int t0 = blockIdx.x * 64;
  int d0 = blockIdx.y * 64;
  const __bf16* src = v + (size_t)b * SEQ * DMODEL + (size_t)h * HDIM;
  __bf16* dst = vt + (size_t)(b * NHEAD + h) * HDIM * SEQ;
#pragma unroll
  for (int i = 0; i < 16; ++i) {
    int lin = (int)threadIdx.x + 256 * i;
    int tt = lin >> 6, dd = lin & 63;                 // coalesced along d
    tile[dd][tt] = src[(size_t)(t0 + tt) * DMODEL + d0 + dd];
  }
  __syncthreads();
#pragma unroll
  for (int i = 0; i < 16; ++i) {
    int lin = (int)threadIdx.x + 256 * i;
    int dd = lin >> 6, tt = lin & 63;                 // coalesced along t
    dst[(size_t)(d0 + dd) * SEQ + t0 + tt] = tile[dd][tt];
  }
}

__global__ void gates_kernel(const float* __restrict__ cv, const float* __restrict__ Wg,
                             const float* __restrict__ bg, float* __restrict__ g) {
  int i = threadIdx.x;
  if (i < BATCH * NHEAD) {
    int b = i / NHEAD, h = i % NHEAD;
    float s = bg[h];
#pragma unroll
    for (int j = 0; j < 10; ++j) s += cv[b * 10 + j] * Wg[h * 10 + j];
    g[i] = 1.f / (1.f + __expf(-s));
  }
}

// RoPE (rotate-half), in place on bf16 Q and K. One thread per (row, head, pair).
__global__ void rope_kernel(__bf16* __restrict__ q, __bf16* __restrict__ k) {
  size_t idx = (size_t)blockIdx.x * blockDim.x + threadIdx.x;
  if (idx >= (size_t)BATCH * SEQ * NHEAD * 64) return;
  int i = (int)(idx & 63);
  int h = (int)((idx >> 6) % NHEAD);
  size_t row = idx / (64 * NHEAD);
  int t = (int)(row & (SEQ - 1));
  float invf = __powf(10000.f, -(float)(2 * i) / 128.f);
  float ang = (float)t * invf;
  float c = __cosf(ang), s = __sinf(ang);
  size_t base = row * DMODEL + (size_t)h * HDIM;
  float a0 = (float)q[base + i], a1 = (float)q[base + i + 64];
  q[base + i]      = (__bf16)(a0 * c - a1 * s);
  q[base + i + 64] = (__bf16)(a1 * c + a0 * s);
  float b0 = (float)k[base + i], b1 = (float)k[base + i + 64];
  k[base + i]      = (__bf16)(b0 * c - b1 * s);
  k[base + i + 64] = (__bf16)(b1 * c + b0 * s);
}

// ---------------------------------------------------------------------------
extern "C" void kernel_launch(void* const* d_in, const int* in_sizes, int n_in,
                              void* d_out, int out_size, void* d_ws, size_t ws_size,
                              hipStream_t stream) {
  (void)in_sizes; (void)n_in; (void)out_size; (void)ws_size;
  const float* x  = (const float*)d_in[0];
  const float* cv = (const float*)d_in[1];
  const float* Wq = (const float*)d_in[2];
  const float* Wk = (const float*)d_in[3];
  const float* Wv = (const float*)d_in[4];
  const float* Wo = (const float*)d_in[5];
  const float* Wg = (const float*)d_in[6];
  const float* bg = (const float*)d_in[7];
  float* out = (float*)d_out;

  const size_t ROWS = (size_t)BATCH * SEQ;    // 16384
  const size_t XE = ROWS * DMODEL;            // 20,971,520 elems
  const size_t WE = (size_t)DMODEL * DMODEL;  // 1,638,400 elems

  __bf16* ws   = (__bf16*)d_ws;
  __bf16* xb   = ws;          // bf16 x; reused as vt after the projections
  __bf16* Wqb  = xb + XE;     // weights kept row-major [e][d], bf16
  __bf16* Wkb  = Wqb + WE;
  __bf16* Wvb  = Wkb + WE;
  __bf16* Wob  = Wvb + WE;
  __bf16* qbuf = Wob + WE;
  __bf16* kbuf = qbuf + XE;
  __bf16* vbuf = kbuf + XE;   // reused as y after the V transpose
  float*  gates = (float*)(vbuf + XE);
  __bf16* vtb = xb;           // alias: x is dead once projections finish
  __bf16* ybuf = vbuf;        // alias: v is dead once transpose finishes

  cvt_bf16_kernel<<<(unsigned)((XE + 255) / 256), 256, 0, stream>>>(x, xb, XE);
  unsigned wb = (unsigned)((WE + 255) / 256);
  cvt_bf16_kernel<<<wb, 256, 0, stream>>>(Wq, Wqb, WE);
  cvt_bf16_kernel<<<wb, 256, 0, stream>>>(Wk, Wkb, WE);
  cvt_bf16_kernel<<<wb, 256, 0, stream>>>(Wv, Wvb, WE);
  cvt_bf16_kernel<<<wb, 256, 0, stream>>>(Wo, Wob, WE);
  gates_kernel<<<1, 128, 0, stream>>>(cv, Wg, bg, gates);

  unsigned gblocks = (unsigned)(((ROWS / 32) * (DMODEL / 64) + 7) / 8);  // 1280
  gemm_bf16_kernel<false><<<gblocks, 256, 0, stream>>>(xb, Wqb, qbuf,
                                                       (int)ROWS, DMODEL, DMODEL);
  gemm_bf16_kernel<false><<<gblocks, 256, 0, stream>>>(xb, Wkb, kbuf,
                                                       (int)ROWS, DMODEL, DMODEL);
  gemm_bf16_kernel<false><<<gblocks, 256, 0, stream>>>(xb, Wvb, vbuf,
                                                       (int)ROWS, DMODEL, DMODEL);

  size_t rn = ROWS * NHEAD * 64;
  rope_kernel<<<(unsigned)((rn + 255) / 256), 256, 0, stream>>>(qbuf, kbuf);

  dim3 tg(SEQ / 64, HDIM / 64, BATCH * NHEAD);
  transpose_v_kernel<<<tg, 256, 0, stream>>>(vbuf, vtb);  // destroys xb (x dead)

  dim3 ag(SEQ / 128, BATCH * NHEAD);
  attn_kernel<<<ag, 256, 0, stream>>>(qbuf, kbuf, vtb, gates, ybuf);

  gemm_bf16_kernel<true><<<gblocks, 256, 0, stream>>>(ybuf, Wob, out,
                                                      (int)ROWS, DMODEL, DMODEL);
}
